// GINEncoderLayerwise_65111704207433
// MI455X (gfx1250) — compile-verified, matrix operand order checked by reading, not run on hardware
//
#include <hip/hip_runtime.h>
#include <hip/hip_bf16.h>

// ---------------------------------------------------------------------------
// GIN encoder, MI455X (gfx1250, wave32, WMMA).
// fp32 scatter-add aggregation (memory/atomic bound, reference-accurate),
// fused MLP per layer via v_wmma_f32_16x16x32_bf16 with:
//   - weights pre-transposed/padded to bf16 once per call (prep pass)
//   - weight tiles staged to LDS via global_load_async_to_lds_b128 (ASYNCcnt)
//   - LDS rows padded by 8 bf16 -> conflict-free ds_load_b128 fragments
//   - software-pipelined K loop: next fragments load before current WMMA
// ---------------------------------------------------------------------------

#define GIN_N      50000
#define GIN_E      800000
#define GIN_DNODE  128
#define GIN_DHID   64
#define GIN_DH2    128
#define GIN_L      5
#define GIN_G      128
#define GIN_EPS    1e-5f
#define GIN_PAD    8     // bf16 elements of row padding (16 bytes)

typedef __attribute__((ext_vector_type(16))) __bf16 v16bf;
typedef __attribute__((ext_vector_type(8)))  float  v8f;

// ---- WMMA 16-bit fragment loader (CDNA5 ISA 7.12.2, wave32) ----------------
// A (16x32, MxK) from row-major [rows x ld]: lanes 0-15 hold M=lane with
// K = {0..7, 16..23}; lanes 16-31 hold M=lane-16 with K = {8..15, 24..31}.
// Same layout serves B when the source is the TRANSPOSED weight (N x K).
// Rows are 16B-aligned (ld % 8 == 0) -> two contiguous ds_load_b128 per lane.
__device__ __forceinline__ v16bf load_frag(const __bf16* __restrict__ s,
                                           int ld, int rowBase, int kBase,
                                           int lane) {
  const int half = lane >> 4;
  const int m    = lane & 15;
  const __bf16* rp = s + (size_t)(rowBase + m) * ld + kBase + half * 8;
  v16bf a;
#pragma unroll
  for (int v = 0; v < 8; ++v) a[v] = rp[v];          // K = half*8 + 0..7
#pragma unroll
  for (int v = 0; v < 8; ++v) a[8 + v] = rp[16 + v]; // K = 16 + half*8 + 0..7
  return a;
}

// ---- prep: W[K x Ncols] fp32 row-major -> bf16 transposed [Ncols x (K+PAD)]
__global__ __launch_bounds__(256) void gin_wprep(__bf16* __restrict__ dst,
                                                 const float* __restrict__ src,
                                                 int K, int Ncols) {
  int gid = blockIdx.x * blockDim.x + threadIdx.x;
  if (gid >= K * Ncols) return;
  int k = gid / Ncols, n = gid - k * Ncols;
  dst[(size_t)n * (K + GIN_PAD) + k] = (__bf16)src[gid];
}

// ---- simple helper kernels -------------------------------------------------
__global__ __launch_bounds__(256) void gin_copy4(float* __restrict__ dst,
                                                 const float* __restrict__ src,
                                                 long n4) {
  long i = (long)blockIdx.x * blockDim.x + threadIdx.x;
  if (i < n4) ((float4*)dst)[i] = ((const float4*)src)[i];
}

__global__ __launch_bounds__(256) void gin_zero(float* __restrict__ p, int n) {
  int i = blockIdx.x * blockDim.x + threadIdx.x;
  if (i < n) p[i] = 0.0f;
}

// Scatter-add: acc[dst,:] += h[src,:]; thread = (edge, 4-feature chunk).
__global__ __launch_bounds__(256) void gin_scatter(float* __restrict__ acc,
                                                   const float* __restrict__ h,
                                                   const int* __restrict__ ei,
                                                   int nEdges, int d) {
  long gid = (long)blockIdx.x * blockDim.x + threadIdx.x;
  const int chunks = d >> 2;
  long e = gid / chunks;
  if (e >= nEdges) return;
  int c4 = (int)(gid - e * chunks) * 4;
  int src = ei[e];
  int dst = ei[(long)nEdges + e];
  const float4 v = *(const float4*)(h + (size_t)src * d + c4);
  float* p = acc + (size_t)dst * d + c4;
  atomicAdd(p + 0, v.x);
  atomicAdd(p + 1, v.y);
  atomicAdd(p + 2, v.z);
  atomicAdd(p + 3, v.w);
}

// ---- fused MLP: relu(bnO(b2 + W2 * relu(bnI(b1 + W1 * a)))) ----------------
// Block: 256 threads = 8 wave32, 64 nodes per block.
// W1T: [128 x (DIN+PAD)] bf16, W2T: [64 x (128+PAD)] bf16 (prepped in global).
template <int DIN>
__global__ __launch_bounds__(256)
void gin_mlp(const float* __restrict__ accIn, int nNodes,
             const __bf16* __restrict__ W1T, const float* __restrict__ b1,
             const float* __restrict__ ig, const float* __restrict__ ibv,
             const float* __restrict__ im, const float* __restrict__ iv,
             const __bf16* __restrict__ W2T, const float* __restrict__ b2,
             const float* __restrict__ og, const float* __restrict__ obv,
             const float* __restrict__ om, const float* __restrict__ ov,
             float* __restrict__ hOut) {
  constexpr int LDA = DIN + GIN_PAD;        // sA / sW1T row stride
  constexpr int LDZ = GIN_DH2 + GIN_PAD;    // sZ / sW2T row stride
  constexpr int KB1 = DIN / 32;             // K blocks, GEMM1
  constexpr int KB2 = GIN_DH2 / 32;         // K blocks, GEMM2
  constexpr int W1B = GIN_DH2 * LDA * 2;    // bytes of W1T
  constexpr int W2B = GIN_DHID * LDZ * 2;   // bytes of W2T

  extern __shared__ __bf16 smem[];
  __bf16* sA   = smem;                      // 64  x LDA
  __bf16* sW1T = sA + 64 * LDA;             // 128 x LDA
  __bf16* sZ   = sW1T + GIN_DH2 * LDA;      // 64  x LDZ
  __bf16* sW2T = sZ + 64 * LDZ;             // 64  x LDZ

  const int tid  = threadIdx.x;
  const int lane = tid & 31;
  const int wave = tid >> 5;
  const int nodeBase = blockIdx.x * 64;

  // Async-copy prepped bf16 weights straight into LDS (16B granules).
  {
    const unsigned ldsW1 = (unsigned)(uintptr_t)sW1T;
    const unsigned ldsW2 = (unsigned)(uintptr_t)sW2T;
    for (int ofs = tid * 16; ofs < W1B; ofs += 256 * 16) {
      asm volatile("global_load_async_to_lds_b128 %0, %1, off"
                   :: "v"(ldsW1 + ofs), "v"((const char*)W1T + ofs)
                   : "memory");
    }
    for (int ofs = tid * 16; ofs < W2B; ofs += 256 * 16) {
      asm volatile("global_load_async_to_lds_b128 %0, %1, off"
                   :: "v"(ldsW2 + ofs), "v"((const char*)W2T + ofs)
                   : "memory");
    }
  }

  // Activation tile: fp32 -> bf16 into padded rows (coalesced reads).
  for (int i = tid; i < 64 * DIN; i += 256) {
    int r = i / DIN, c = i - r * DIN;
    int node = nodeBase + r;
    sA[r * LDA + c] = (node < nNodes) ? (__bf16)accIn[(size_t)node * DIN + c]
                                      : (__bf16)0.0f;
  }

  asm volatile("s_wait_asynccnt 0" ::: "memory");
  __syncthreads();

  const int half = lane >> 4;
  const int lm   = lane & 15;

  // GEMM1: z[64x128] = A[64xDIN] * W1[DINx128]; 4x8 tiles over 8 waves.
  // K loop software-pipelined: next fragments issue before current WMMA.
  for (int t = wave; t < 32; t += 8) {
    const int rt = t >> 3, ct = t & 7;
    v8f acc = {};
    v16bf a = load_frag(sA, LDA, rt * 16, 0, lane);
    v16bf b = load_frag(sW1T, LDA, ct * 16, 0, lane);  // W1^T: N x K
#pragma unroll
    for (int kb = 0; kb < KB1; ++kb) {
      v16bf an = a, bn = b;
      if (kb + 1 < KB1) {
        an = load_frag(sA, LDA, rt * 16, (kb + 1) * 32, lane);
        bn = load_frag(sW1T, LDA, ct * 16, (kb + 1) * 32, lane);
      }
      acc = __builtin_amdgcn_wmma_f32_16x16x32_bf16(
          false, a, false, b, (short)0, acc, false, false);
      a = an; b = bn;
    }
    const int col = ct * 16 + lm;
    const float s  = ig[col] * rsqrtf(iv[col] + GIN_EPS);
    const float sh = fmaf(b1[col] - im[col], s, ibv[col]);
#pragma unroll
    for (int j = 0; j < 8; ++j) {
      const int row = rt * 16 + half * 8 + j;  // D layout: M = vgpr + 8*half
      float z = fmaxf(fmaf(acc[j], s, sh), 0.0f);
      sZ[(size_t)row * LDZ + col] = (__bf16)z;
    }
  }
  __syncthreads();

  // GEMM2: out[64x64] = Z[64x128] * W2[128x64]; 4x4 tiles over 8 waves.
  for (int t = wave; t < 16; t += 8) {
    const int rt = t >> 2, ct = t & 3;
    v8f acc = {};
    v16bf a = load_frag(sZ, LDZ, rt * 16, 0, lane);
    v16bf b = load_frag(sW2T, LDZ, ct * 16, 0, lane);  // W2^T: N x K
#pragma unroll
    for (int kb = 0; kb < KB2; ++kb) {
      v16bf an = a, bn = b;
      if (kb + 1 < KB2) {
        an = load_frag(sZ, LDZ, rt * 16, (kb + 1) * 32, lane);
        bn = load_frag(sW2T, LDZ, ct * 16, (kb + 1) * 32, lane);
      }
      acc = __builtin_amdgcn_wmma_f32_16x16x32_bf16(
          false, a, false, b, (short)0, acc, false, false);
      a = an; b = bn;
    }
    const int col = ct * 16 + lm;
    const float s  = og[col] * rsqrtf(ov[col] + GIN_EPS);
    const float sh = fmaf(b2[col] - om[col], s, obv[col]);
#pragma unroll
    for (int j = 0; j < 8; ++j) {
      const int row  = rt * 16 + half * 8 + j;
      const int node = nodeBase + row;
      float o = fmaxf(fmaf(acc[j], s, sh), 0.0f);
      if (node < nNodes) hOut[(size_t)node * GIN_DHID + col] = o;
    }
  }
}

// ---- readout ---------------------------------------------------------------
__global__ __launch_bounds__(256) void gin_pool(const float* __restrict__ h,
                                                const int* __restrict__ batch,
                                                float* __restrict__ sums,
                                                float* __restrict__ cnts,
                                                int nNodes) {
  long gid = (long)blockIdx.x * blockDim.x + threadIdx.x;
  if (gid >= (long)nNodes * GIN_DHID) return;
  int n = (int)(gid >> 6), c = (int)(gid & 63);
  int g = batch[n];
  atomicAdd(&sums[(size_t)g * GIN_DHID + c], h[(size_t)n * GIN_DHID + c]);
  if (c == 0) atomicAdd(&cnts[g], 1.0f);
}

__global__ __launch_bounds__(256) void gin_pool_fin(const float* __restrict__ sums,
                                                    const float* __restrict__ cnts,
                                                    float* __restrict__ out) {
  int gid = blockIdx.x * blockDim.x + threadIdx.x;
  if (gid >= GIN_G * GIN_DHID) return;
  out[gid] = sums[gid] / fmaxf(cnts[gid >> 6], 1.0f);
}

// ---------------------------------------------------------------------------
extern "C" void kernel_launch(void* const* d_in, const int* in_sizes, int n_in,
                              void* d_out, int out_size, void* d_ws, size_t ws_size,
                              hipStream_t stream) {
  const float* x     = (const float*)d_in[0];
  const int*   ei    = (const int*)d_in[1];
  const int*   batch = (const int*)d_in[2];
  const float* w1_0  = (const float*)d_in[3];
  const float* b1_0  = (const float*)d_in[4];
  const float* ig0   = (const float*)d_in[5];
  const float* ib0   = (const float*)d_in[6];
  const float* im0   = (const float*)d_in[7];
  const float* iv0   = (const float*)d_in[8];
  const float* w2_0  = (const float*)d_in[9];
  const float* b2_0  = (const float*)d_in[10];
  const float* W1s   = (const float*)d_in[11];
  const float* b1s   = (const float*)d_in[12];
  const float* IG    = (const float*)d_in[13];
  const float* IB    = (const float*)d_in[14];
  const float* IM    = (const float*)d_in[15];
  const float* IV    = (const float*)d_in[16];
  const float* W2s   = (const float*)d_in[17];
  const float* b2s   = (const float*)d_in[18];
  const float* OG    = (const float*)d_in[19];
  const float* OB    = (const float*)d_in[20];
  const float* OM    = (const float*)d_in[21];
  const float* OV    = (const float*)d_in[22];

  // Workspace: fp32 region | bf16 prepped-weights region.
  float* ws   = (float*)d_ws;
  float* acc  = ws;                                   // N*128
  float* hA   = acc + (size_t)GIN_N * GIN_DNODE;      // N*64
  float* hB   = hA + (size_t)GIN_N * GIN_DHID;        // N*64
  float* sums = hB + (size_t)GIN_N * GIN_DHID;        // G*64
  float* cnts = sums + GIN_G * GIN_DHID;              // G
  __bf16* wb  = (__bf16*)(cnts + GIN_G);              // 16B aligned
  // bf16 layout: W1T0[128*136] | W1Tm[4][128*72] | W2T[5][64*136]
  const size_t szW1T0 = (size_t)GIN_DH2 * (GIN_DNODE + GIN_PAD);  // 17408
  const size_t szW1Tm = (size_t)GIN_DH2 * (GIN_DHID + GIN_PAD);   //  9216
  const size_t szW2T  = (size_t)GIN_DHID * (GIN_DH2 + GIN_PAD);   //  8704
  __bf16* W1T0 = wb;
  __bf16* W1Tm = W1T0 + szW1T0;
  __bf16* W2Tl = W1Tm + 4 * szW1Tm;

  // ---- prep pass: transpose+convert all weights to bf16 ----
  gin_wprep<<<(GIN_DNODE * GIN_DH2 + 255) / 256, 256, 0, stream>>>(
      W1T0, w1_0, GIN_DNODE, GIN_DH2);
  for (int m = 0; m < 4; ++m)
    gin_wprep<<<(GIN_DHID * GIN_DH2 + 255) / 256, 256, 0, stream>>>(
        W1Tm + m * szW1Tm, W1s + (size_t)m * GIN_DHID * GIN_DH2,
        GIN_DHID, GIN_DH2);
  gin_wprep<<<(GIN_DH2 * GIN_DHID + 255) / 256, 256, 0, stream>>>(
      W2Tl, w2_0, GIN_DH2, GIN_DHID);
  for (int m = 0; m < 4; ++m)
    gin_wprep<<<(GIN_DH2 * GIN_DHID + 255) / 256, 256, 0, stream>>>(
        W2Tl + (m + 1) * szW2T, W2s + (size_t)m * GIN_DH2 * GIN_DHID,
        GIN_DH2, GIN_DHID);

  const int mlpGrid = (GIN_N + 63) / 64;
  const size_t lds128 =
      (size_t)(64 + 128) * (GIN_DNODE + GIN_PAD) * 2 +
      (size_t)(64 + 64) * (GIN_DH2 + GIN_PAD) * 2;    // 87040 B
  const size_t lds64 =
      (size_t)(64 + 128) * (GIN_DHID + GIN_PAD) * 2 +
      (size_t)(64 + 64) * (GIN_DH2 + GIN_PAD) * 2;    // 62464 B

  // ---- layer 0 (input dim 128) ----
  {
    long n4 = (long)GIN_N * GIN_DNODE / 4;
    gin_copy4<<<(int)((n4 + 255) / 256), 256, 0, stream>>>(acc, x, n4);
    long sthreads = (long)GIN_E * (GIN_DNODE / 4);
    gin_scatter<<<(int)((sthreads + 255) / 256), 256, 0, stream>>>(
        acc, x, ei, GIN_E, GIN_DNODE);
    gin_mlp<GIN_DNODE><<<mlpGrid, 256, lds128, stream>>>(
        acc, GIN_N, W1T0, b1_0, ig0, ib0, im0, iv0, W2Tl, b2_0,
        OG + 0, OB + 0, OM + 0, OV + 0, hA);
  }

  // ---- layers 1..4 (input dim 64) ----
  float* cur = hA;
  float* nxt = hB;
  for (int l = 1; l < GIN_L; ++l) {
    const int m = l - 1;
    long n4 = (long)GIN_N * GIN_DHID / 4;
    gin_copy4<<<(int)((n4 + 255) / 256), 256, 0, stream>>>(acc, cur, n4);
    long sthreads = (long)GIN_E * (GIN_DHID / 4);
    gin_scatter<<<(int)((sthreads + 255) / 256), 256, 0, stream>>>(
        acc, cur, ei, GIN_E, GIN_DHID);
    gin_mlp<GIN_DHID><<<mlpGrid, 256, lds64, stream>>>(
        acc, GIN_N,
        W1Tm + m * szW1Tm, b1s + (size_t)m * GIN_DH2,
        IG + (size_t)m * GIN_DH2, IB + (size_t)m * GIN_DH2,
        IM + (size_t)m * GIN_DH2, IV + (size_t)m * GIN_DH2,
        W2Tl + (size_t)l * szW2T, b2s + (size_t)m * GIN_DHID,
        OG + (size_t)l * GIN_DHID, OB + (size_t)l * GIN_DHID,
        OM + (size_t)l * GIN_DHID, OV + (size_t)l * GIN_DHID, nxt);
    float* tmp = cur; cur = nxt; nxt = tmp;
  }

  // ---- readout: segment mean over graphs ----
  gin_zero<<<(GIN_G * GIN_DHID + GIN_G + 255) / 256, 256, 0, stream>>>(
      sums, GIN_G * GIN_DHID + GIN_G);
  long pthreads = (long)GIN_N * GIN_DHID;
  gin_pool<<<(int)((pthreads + 255) / 256), 256, 0, stream>>>(
      cur, batch, sums, cnts, GIN_N);
  gin_pool_fin<<<(GIN_G * GIN_DHID + 255) / 256, 256, 0, stream>>>(
      sums, cnts, (float*)d_out);
}
